// GInRecSimple_2413771620669
// MI455X (gfx1250) — compile-verified, matrix operand order checked by reading, not run on hardware
//
#include <hip/hip_runtime.h>
#include <hip/hip_bf16.h>

#define N_ENT 50000
#define N_USR 10000
#define NN    60000          // N_ENT + N_USR, divisible by 16 (3750 tiles)

typedef __attribute__((ext_vector_type(16))) __bf16 v16bf;
typedef __attribute__((ext_vector_type(8)))  __bf16 v8bf;
typedef __attribute__((ext_vector_type(4)))  __bf16 v4bf;
typedef __attribute__((ext_vector_type(8)))  float  v8f;

// ---------------------------------------------------------------------------
// CDNA5 WMMA 16-bit layouts (wave32):
//   A 16x32 (MxK): lane -> M = lane&15, half = lane>>4
//     VGPR v holds K = (v<4?0:16) + (half?8:0) + 2*(v&3)  and K+1
//   B 32x16 (KxN): symmetric (lane -> N), same K pattern
//   C/D 16x16 f32: VGPR v holds row M = v + 8*half, col N = lane&15
// Consequence: per lane an A fragment is two contiguous 16-byte runs:
//   [base, base+8) and [base+16, base+24) bf16 elems, base = kt*32 + hf*8
// ---------------------------------------------------------------------------
static __device__ __forceinline__ int frag_k(int v, int hf) {
    return (v < 4 ? 0 : 16) + (hf ? 8 : 0) + 2 * (v & 3);
}

// A fragment from a bf16 LDS tile [16][ld]: 2x ds_load_b128 per lane
static __device__ __forceinline__ v16bf frag_a_ldsb(const __bf16* sh, int ld, int kt, int lane) {
    int m = lane & 15, hf = (lane >> 4) & 1;
    const __bf16* p = sh + m * ld + kt * 32 + hf * 8;
    v8bf lo = *(const v8bf*)p;
    v8bf hi = *(const v8bf*)(p + 16);
    return __builtin_shufflevector(lo, hi, 0, 1, 2, 3, 4, 5, 6, 7,
                                           8, 9, 10, 11, 12, 13, 14, 15);
}

// B fragment from a pre-packed fragment buffer: one 32-byte load per lane
static __device__ __forceinline__ v16bf frag_b_pk(const v16bf* P, int NT, int nt, int kt, int lane) {
    return P[(kt * NT + nt) * 32 + lane];
}

static __device__ __forceinline__ v8f wmma_bf16(v16bf a, v16bf b, v8f c) {
    return __builtin_amdgcn_wmma_f32_16x16x32_bf16(false, a, false, b, (short)0, c, false, false);
}

// ---------------------------------------------------------------------------
// Prep kernels
// ---------------------------------------------------------------------------
__global__ void k_zero(float* p, int n) {
    for (int i = blockIdx.x * blockDim.x + threadIdx.x; i < n; i += gridDim.x * blockDim.x)
        p[i] = 0.0f;
}

// Pack W [K][N] f32 (row-major) into per-(kt,nt,lane) v16bf fragments
__global__ void k_pack_b(const float* __restrict__ W, __bf16* __restrict__ P, int K, int N) {
    int idx = blockIdx.x * blockDim.x + threadIdx.x;
    if (idx >= K * N) return;
    int NT   = N >> 4;
    int tile = idx >> 9;            // 512 elems per 16x16x32 fragment pair
    int rem  = idx & 511;
    int lane = rem >> 4;
    int e    = rem & 15;
    int kt = tile / NT, nt = tile % NT;
    int v = e >> 1, pbit = e & 1, hf = lane >> 4;
    int n = nt * 16 + (lane & 15);
    int k = kt * 32 + frag_k(v, hf) + pbit;
    P[idx] = (__bf16)W[k * N + n];
}

// w_sum[r*64+j] = sum_k Wr[r][j][k], Wr row-major [R][64][32]
__global__ void k_wsum(const float* __restrict__ Wr, float* __restrict__ ws, int total) {
    int i = blockIdx.x * blockDim.x + threadIdx.x;
    if (i >= total) return;
    const float* p = Wr + (size_t)i * 32;
    float s = 0.0f;
#pragma unroll
    for (int k = 0; k < 32; ++k) s += p[k];
    ws[i] = s;
}

__global__ void k_degree(const int* __restrict__ dst, int* __restrict__ cnt, int e) {
    int i = blockIdx.x * blockDim.x + threadIdx.x;
    if (i < e) atomicAdd(&cnt[dst[i]], 1);
}

// ---------------------------------------------------------------------------
// Autoencoder: enc = relu(x@W0+b0)@W1+b1 ; dec = relu(enc@dW0+db0)@dW1+db1
// one wave per 16-node tile; accumulates MSE loss
// ---------------------------------------------------------------------------
__global__ __launch_bounds__(32) void k_autoencoder(
    const float* __restrict__ ent, const float* __restrict__ usr,
    const v16bf* __restrict__ W0p, const float* __restrict__ b0,
    const v16bf* __restrict__ W1p, const float* __restrict__ b1,
    const v16bf* __restrict__ dW0p, const float* __restrict__ db0,
    const v16bf* __restrict__ dW1p, const float* __restrict__ db1,
    float* __restrict__ encoded, float* __restrict__ loss_out)
{
    __shared__ __align__(16) float  shx [16 * 64];   // f32 copy for the loss
    __shared__ __align__(16) __bf16 shxb[16 * 64];
    __shared__ __align__(16) __bf16 shh [16 * 128];
    __shared__ __align__(16) __bf16 she [16 * 32];
    const int lane = threadIdx.x;
    const int m0 = blockIdx.x * 16;
    const int n_l = lane & 15, hf = lane >> 4;

    // load input tile as float4 (8 x b128 per lane), mirror to bf16
    for (int i = lane; i < 256; i += 32) {
        int r = i >> 4, c4 = i & 15, g = m0 + r;
        const float4* sp = (g < N_ENT) ? (const float4*)(ent + (size_t)g * 64)
                                       : (const float4*)(usr + (size_t)(g - N_ENT) * 64);
        float4 vv = sp[c4];
        *(float4*)(shx + r * 64 + c4 * 4) = vv;
        v4bf bv;
        bv[0] = (__bf16)vv.x; bv[1] = (__bf16)vv.y;
        bv[2] = (__bf16)vv.z; bv[3] = (__bf16)vv.w;
        *(v4bf*)(shxb + r * 64 + c4 * 4) = bv;
    }
    __syncthreads();

    // h1 = relu(x @ W0 + b0)   [16,128], K=64
#pragma unroll
    for (int nt = 0; nt < 8; ++nt) {
        v8f c; float bb = b0[nt * 16 + n_l];
#pragma unroll
        for (int v = 0; v < 8; ++v) c[v] = bb;
#pragma unroll
        for (int kt = 0; kt < 2; ++kt)
            c = wmma_bf16(frag_a_ldsb(shxb, 64, kt, lane), frag_b_pk(W0p, 8, nt, kt, lane), c);
#pragma unroll
        for (int v = 0; v < 8; ++v) {
            float t = c[v];
            shh[(v + hf * 8) * 128 + nt * 16 + n_l] = (__bf16)(t > 0.0f ? t : 0.0f);
        }
    }
    __syncthreads();

    // enc = h1 @ W1 + b1   [16,32], K=128 ; write f32 to global + bf16 to LDS
#pragma unroll
    for (int nt = 0; nt < 2; ++nt) {
        v8f c; float bb = b1[nt * 16 + n_l];
#pragma unroll
        for (int v = 0; v < 8; ++v) c[v] = bb;
#pragma unroll
        for (int kt = 0; kt < 4; ++kt)
            c = wmma_bf16(frag_a_ldsb(shh, 128, kt, lane), frag_b_pk(W1p, 2, nt, kt, lane), c);
#pragma unroll
        for (int v = 0; v < 8; ++v) {
            float t = c[v];
            encoded[(size_t)(m0 + v + hf * 8) * 32 + nt * 16 + n_l] = t;
            she[(v + hf * 8) * 32 + nt * 16 + n_l] = (__bf16)t;
        }
    }
    __syncthreads();

    // h2 = relu(enc @ dW0 + db0)   [16,128], K=32
#pragma unroll
    for (int nt = 0; nt < 8; ++nt) {
        v8f c; float bb = db0[nt * 16 + n_l];
#pragma unroll
        for (int v = 0; v < 8; ++v) c[v] = bb;
        c = wmma_bf16(frag_a_ldsb(she, 32, 0, lane), frag_b_pk(dW0p, 8, nt, 0, lane), c);
#pragma unroll
        for (int v = 0; v < 8; ++v) {
            float t = c[v];
            shh[(v + hf * 8) * 128 + nt * 16 + n_l] = (__bf16)(t > 0.0f ? t : 0.0f);
        }
    }
    __syncthreads();

    // dec = h2 @ dW1 + db1   [16,64], K=128 ; fold in MSE loss vs f32 input
    float ls = 0.0f;
#pragma unroll
    for (int nt = 0; nt < 4; ++nt) {
        v8f c; float bb = db1[nt * 16 + n_l];
#pragma unroll
        for (int v = 0; v < 8; ++v) c[v] = bb;
#pragma unroll
        for (int kt = 0; kt < 4; ++kt)
            c = wmma_bf16(frag_a_ldsb(shh, 128, kt, lane), frag_b_pk(dW1p, 4, nt, kt, lane), c);
#pragma unroll
        for (int v = 0; v < 8; ++v) {
            float d = c[v] - shx[(v + hf * 8) * 64 + nt * 16 + n_l];
            ls += d * d;
        }
    }
#pragma unroll
    for (int off = 16; off > 0; off >>= 1) ls += __shfl_xor(ls, off, 32);
    if (lane == 0) atomicAdd(loss_out, ls * (1.0f / (60000.0f * 64.0f)));
}

// ---------------------------------------------------------------------------
// Gated edge aggregation: one edge per wave, lane = feature (dim 32)
// ---------------------------------------------------------------------------
__global__ void k_gate_agg(const float* __restrict__ x,
                           const int* __restrict__ esrc, const int* __restrict__ edst,
                           const int* __restrict__ erel,
                           const float* __restrict__ wsum, float* __restrict__ agg, int e_cnt)
{
    int w = blockIdx.x * (blockDim.x >> 5) + (threadIdx.x >> 5);
    int lane = threadIdx.x & 31;
    if (w >= e_cnt) return;
    w = __builtin_amdgcn_readfirstlane(w);             // force SGPR -> scalar index loads
    int s = esrc[w], d = edst[w], r = erel[w];
    float hd = x[(size_t)d * 32 + lane];
    float hs = x[(size_t)s * 32 + lane];
    float part = hd * wsum[r * 64 + lane] + hs * wsum[r * 64 + 32 + lane];
#pragma unroll
    for (int off = 16; off > 0; off >>= 1) part += __shfl_xor(part, off, 32);
    float gate = 1.0f / (1.0f + expf(-part));
    atomicAdd(&agg[(size_t)d * 32 + lane], hs * gate);
}

// ---------------------------------------------------------------------------
// Combine: out = leaky_relu(concat(x, mean_agg) @ linW + b)  via WMMA
// ---------------------------------------------------------------------------
__global__ __launch_bounds__(32) void k_combine(
    const float* __restrict__ x, const float* __restrict__ agg, const int* __restrict__ cnt,
    const v16bf* __restrict__ linWp, const float* __restrict__ linb,
    float* __restrict__ out, int outNT)
{
    __shared__ __align__(16) __bf16 shc[16 * 64];
    const int lane = threadIdx.x;
    const int m0 = blockIdx.x * 16;
    const int n_l = lane & 15, hf = lane >> 4;
    const int outdim = outNT << 4;

    // build tile: cols 0-31 = x[g], cols 32-63 = agg[g]/cnt[g] (float4 granules)
    for (int i = lane; i < 256; i += 32) {
        int r = i >> 4, c4 = i & 15, g = m0 + r;
        float4 vv;
        if (c4 < 8) {
            vv = ((const float4*)(x + (size_t)g * 32))[c4];
        } else {
            float4 s = ((const float4*)(agg + (size_t)g * 32))[c4 - 8];
            int cc = cnt[g];
            float inv = (cc > 0) ? 1.0f / (float)cc : 0.0f;
            vv = make_float4(s.x * inv, s.y * inv, s.z * inv, s.w * inv);
        }
        v4bf bv;
        bv[0] = (__bf16)vv.x; bv[1] = (__bf16)vv.y;
        bv[2] = (__bf16)vv.z; bv[3] = (__bf16)vv.w;
        *(v4bf*)(shc + r * 64 + c4 * 4) = bv;
    }
    __syncthreads();

    for (int nt = 0; nt < outNT; ++nt) {
        v8f c; float bb = linb[nt * 16 + n_l];
#pragma unroll
        for (int v = 0; v < 8; ++v) c[v] = bb;
#pragma unroll
        for (int kt = 0; kt < 2; ++kt)
            c = wmma_bf16(frag_a_ldsb(shc, 64, kt, lane), frag_b_pk(linWp, outNT, nt, kt, lane), c);
#pragma unroll
        for (int v = 0; v < 8; ++v) {
            float t = c[v];
            t = t > 0.0f ? t : 0.01f * t;  // leaky relu
            out[(size_t)(m0 + v + hf * 8) * outdim + nt * 16 + n_l] = t;
        }
    }
}

// ---------------------------------------------------------------------------
// Scoring: score[b] = dot(final[user], final[item]), final = concat(x1, x2)
// ---------------------------------------------------------------------------
__global__ void k_score(const int* __restrict__ users, const int* __restrict__ items,
                        const float* __restrict__ x1, const float* __restrict__ x2,
                        float* __restrict__ out, int b_cnt)
{
    int t = blockIdx.x * blockDim.x + threadIdx.x;
    if (t >= b_cnt) return;
    size_t u  = (size_t)(N_ENT + users[t]);
    size_t it = (size_t)items[t];
    const float4* a1 = (const float4*)(x1 + u * 32);
    const float4* b1 = (const float4*)(x1 + it * 32);
    const float4* a2 = (const float4*)(x2 + u * 16);
    const float4* b2 = (const float4*)(x2 + it * 16);
    float s = 0.0f;
#pragma unroll
    for (int j = 0; j < 8; ++j) {
        float4 a = a1[j], b = b1[j];
        s += a.x * b.x + a.y * b.y + a.z * b.z + a.w * b.w;
    }
#pragma unroll
    for (int j = 0; j < 4; ++j) {
        float4 a = a2[j], b = b2[j];
        s += a.x * b.x + a.y * b.y + a.z * b.z + a.w * b.w;
    }
    out[t] = s;
}

// ---------------------------------------------------------------------------
// Launch
// ---------------------------------------------------------------------------
extern "C" void kernel_launch(void* const* d_in, const int* in_sizes, int n_in,
                              void* d_out, int out_size, void* d_ws, size_t ws_size,
                              hipStream_t stream) {
    const int*   users   = (const int*)d_in[0];
    const int*   items   = (const int*)d_in[1];
    const int*   esrc    = (const int*)d_in[2];
    const int*   edst    = (const int*)d_in[3];
    const int*   erel    = (const int*)d_in[4];
    const float* ent     = (const float*)d_in[5];
    const float* usr     = (const float*)d_in[6];
    const float* encW0   = (const float*)d_in[7];
    const float* encb0   = (const float*)d_in[8];
    const float* encW1   = (const float*)d_in[9];
    const float* encb1   = (const float*)d_in[10];
    const float* decW0   = (const float*)d_in[11];
    const float* decb0   = (const float*)d_in[12];
    const float* decW1   = (const float*)d_in[13];
    const float* decb1   = (const float*)d_in[14];
    const float* Wr1     = (const float*)d_in[15];
    const float* linW1   = (const float*)d_in[16];
    const float* linb1   = (const float*)d_in[17];
    const float* Wr2     = (const float*)d_in[18];
    const float* linW2   = (const float*)d_in[19];
    const float* linb2   = (const float*)d_in[20];

    const int E = in_sizes[2];
    const int B = in_sizes[0];

    // workspace carve
    char* ws = (char*)d_ws;
    size_t off = 0;
    auto carve = [&](size_t bytes) -> void* {
        void* p = ws + off;
        off = (off + bytes + 255) & ~(size_t)255;
        return p;
    };
    float*  encoded = (float*)carve((size_t)NN * 32 * sizeof(float));
    float*  x1      = (float*)carve((size_t)NN * 32 * sizeof(float));
    float*  x2      = (float*)carve((size_t)NN * 16 * sizeof(float));
    float*  agg     = (float*)carve((size_t)NN * 32 * sizeof(float));
    int*    cnt     = (int*)  carve((size_t)NN * sizeof(int));
    float*  wsum1   = (float*)carve(16 * 64 * sizeof(float));
    float*  wsum2   = (float*)carve(16 * 64 * sizeof(float));
    __bf16* pW0     = (__bf16*)carve(64 * 128 * sizeof(__bf16));
    __bf16* pW1     = (__bf16*)carve(128 * 32 * sizeof(__bf16));
    __bf16* pdW0    = (__bf16*)carve(32 * 128 * sizeof(__bf16));
    __bf16* pdW1    = (__bf16*)carve(128 * 64 * sizeof(__bf16));
    __bf16* plin1   = (__bf16*)carve(64 * 32 * sizeof(__bf16));
    __bf16* plin2   = (__bf16*)carve(64 * 16 * sizeof(__bf16));

    float* out   = (float*)d_out;        // [0..B-1] scores, [B] ae_loss
    float* lossp = out + B;

    // --- prep: pack weights into WMMA B-fragment layout + fold gate weights ---
    k_pack_b<<<(64 * 128 + 255) / 256, 256, 0, stream>>>(encW0, pW0, 64, 128);
    k_pack_b<<<(128 * 32 + 255) / 256, 256, 0, stream>>>(encW1, pW1, 128, 32);
    k_pack_b<<<(32 * 128 + 255) / 256, 256, 0, stream>>>(decW0, pdW0, 32, 128);
    k_pack_b<<<(128 * 64 + 255) / 256, 256, 0, stream>>>(decW1, pdW1, 128, 64);
    k_pack_b<<<(64 * 32 + 255) / 256, 256, 0, stream>>>(linW1, plin1, 64, 32);
    k_pack_b<<<(64 * 16 + 255) / 256, 256, 0, stream>>>(linW2, plin2, 64, 16);
    k_wsum<<<(16 * 64 + 255) / 256, 256, 0, stream>>>(Wr1, wsum1, 16 * 64);
    k_wsum<<<(16 * 64 + 255) / 256, 256, 0, stream>>>(Wr2, wsum2, 16 * 64);

    // --- init accumulators (in-stream, deterministic each call) ---
    k_zero<<<512, 256, 0, stream>>>((float*)cnt, NN);
    k_zero<<<512, 256, 0, stream>>>(agg, NN * 32);
    k_zero<<<1, 32, 0, stream>>>(lossp, 1);

    // --- degree counts (shared by both conv layers) ---
    k_degree<<<(E + 255) / 256, 256, 0, stream>>>(edst, cnt, E);

    // --- autoencoder (WMMA) ---
    k_autoencoder<<<NN / 16, 32, 0, stream>>>(
        ent, usr, (const v16bf*)pW0, encb0, (const v16bf*)pW1, encb1,
        (const v16bf*)pdW0, decb0, (const v16bf*)pdW1, decb1, encoded, lossp);

    // --- conv layer 1 ---
    k_gate_agg<<<(E + 7) / 8, 256, 0, stream>>>(encoded, esrc, edst, erel, wsum1, agg, E);
    k_combine<<<NN / 16, 32, 0, stream>>>(encoded, agg, cnt, (const v16bf*)plin1, linb1, x1, 2);

    // --- conv layer 2 (re-zero accumulator first) ---
    k_zero<<<512, 256, 0, stream>>>(agg, NN * 32);
    k_gate_agg<<<(E + 7) / 8, 256, 0, stream>>>(x1, esrc, edst, erel, wsum2, agg, E);
    k_combine<<<NN / 16, 32, 0, stream>>>(x1, agg, cnt, (const v16bf*)plin2, linb2, x2, 1);

    // --- scoring ---
    k_score<<<(B + 255) / 256, 256, 0, stream>>>(users, items, x1, x2, out, B);
}
